// AbstractAttention_80942953660906
// MI455X (gfx1250) — matrix operand order repaired and checked by reading, not compiled, not run on hardware
//
#include <hip/hip_runtime.h>
#include <stdint.h>

// Problem constants (from reference)
constexpr int kB  = 2;
constexpr int kS  = 2048;
constexpr int kD  = 4096;
constexpr int kH  = 32;
constexpr int kHD = 128;   // head dim

typedef __bf16 bf16_t;
typedef __attribute__((ext_vector_type(16))) __bf16 v16bf;
typedef __attribute__((ext_vector_type(8)))  float  v8f;
typedef __attribute__((ext_vector_type(4)))  int    v4i;

union FragBF {
    v16bf v;
    v4i   q[2];
};

__device__ __forceinline__ v8f wmma_bf16(v16bf a, v16bf b, v8f c) {
    // (neg_a, A, neg_b, B, c_mod, C, reuse_a, reuse_b)
    return __builtin_amdgcn_wmma_f32_16x16x32_bf16(false, a, false, b, (short)0, c,
                                                   false, false);
}

// A-matrix fragment (16x32 bf16, row-major source).
// Lanes 0-15: K = [0,8) and [16,24); lanes 16-31: K = [8,16) and [24,32).
__device__ __forceinline__ v16bf load_frag_a(const bf16_t* rowptr, int k0, int hx) {
    FragBF f;
    f.q[0] = *(const v4i*)(rowptr + k0 + hx * 8);
    f.q[1] = *(const v4i*)(rowptr + k0 + 16 + hx * 8);
    return f.v;
}

// B-matrix fragment (32x16 bf16). colptr = start of this lane's column,
// contiguous along K. Lanes 0-15: K = [0,8) and [16,24); lanes 16-31: K = [8,16) and [24,32).
__device__ __forceinline__ v16bf load_frag_b(const bf16_t* colptr, int k0, int hx) {
    FragBF f;
    const bf16_t* p = colptr + k0 + hx * 8;
    f.q[0] = *(const v4i*)(p);
    f.q[1] = *(const v4i*)(p + 16);
    return f.v;
}

// CDNA5 async DMA: global -> LDS, 16 bytes per lane, tracked by ASYNCcnt.
__device__ __forceinline__ void async_ld_b128(unsigned lds_off, uint64_t gaddr) {
    asm volatile("global_load_async_to_lds_b128 %0, %1, off"
                 :: "v"(lds_off), "v"(gaddr)
                 : "memory");
}
__device__ __forceinline__ void wait_async0() {
    asm volatile("s_wait_asynccnt 0" ::: "memory");
}

// ---------------------------------------------------------------------------
// f32 -> bf16 conversion
// ---------------------------------------------------------------------------
__global__ void cvt_f32_bf16(const float* __restrict__ src,
                             bf16_t* __restrict__ dst, int n) {
    int i = blockIdx.x * blockDim.x + threadIdx.x;
    if (i < n) dst[i] = (bf16_t)src[i];
}

// ---------------------------------------------------------------------------
// GEMM: out[m,n] = sum_k A[m,k] * W[n,k]   (A row-major MxK, W row-major NxK)
// Block = 256 threads = 8 waves computing a 128x128 tile.
// Per K-chunk (64): async-DMA A(128x64) + B(128x64) into LDS (32 KB),
// then each wave (32x64 sub-tile, 2x4 C frags) runs 16 WMMAs off LDS.
// EPI 0: store bf16 transposed per head -> Vt[b,h,hd,s]
// EPI 1: RoPE, store bf16 head-major    -> Qh/Kh[b,h,s,hd]
// EPI 2: store f32 row-major            -> out32[m*N+n]
// ---------------------------------------------------------------------------
template <int EPI>
__global__ __launch_bounds__(256) void gemm_bf16_kernel(
    const bf16_t* __restrict__ A, const bf16_t* __restrict__ Wb,
    bf16_t* __restrict__ out16, float* __restrict__ out32,
    const float* __restrict__ cosT, const float* __restrict__ sinT,
    int M, int N, int K) {
    __shared__ __align__(16) bf16_t Asl[128 * 64];   // [row][k] row stride 64
    __shared__ __align__(16) bf16_t Bsl[128 * 64];   // [col][k] col stride 64

    const int t    = threadIdx.x;
    const int lane = t & 31;
    const int w    = t >> 5;
    const int hx   = lane >> 4;
    const int l16  = lane & 15;
    const int wm   = w >> 1;  // 0..3
    const int wn   = w & 1;   // 0..1
    const int rowBase = blockIdx.y * 128;
    const int colBase = blockIdx.x * 128;
    const int tm = rowBase + wm * 32;
    const int tn = colBase + wn * 64;

    v8f c[2][4] = {};

    for (int k0 = 0; k0 < K; k0 += 64) {
        __syncthreads();  // previous chunk's LDS reads done before refill
        // ---- stage A: 1024 16B chunks, 4 per thread ----
#pragma unroll
        for (int i = 0; i < 4; ++i) {
            int cidx = t + i * 256;               // 0..1023
            int row  = cidx >> 3;
            int kp   = (cidx & 7) * 8;
            uint64_t g = (uint64_t)(uintptr_t)(A + (size_t)(rowBase + row) * K + k0 + kp);
            async_ld_b128((unsigned)(uintptr_t)&Asl[row * 64 + kp], g);
        }
        // ---- stage B: 1024 16B chunks, 4 per thread ----
#pragma unroll
        for (int i = 0; i < 4; ++i) {
            int cidx = t + i * 256;
            int col  = cidx >> 3;
            int kp   = (cidx & 7) * 8;
            uint64_t g = (uint64_t)(uintptr_t)(Wb + (size_t)(colBase + col) * K + k0 + kp);
            async_ld_b128((unsigned)(uintptr_t)&Bsl[col * 64 + kp], g);
        }
        wait_async0();
        __syncthreads();

        // ---- compute off LDS: 2 K-steps of 32 ----
#pragma unroll
        for (int ks = 0; ks < 2; ++ks) {
            v16bf af[2], bfr[4];
#pragma unroll
            for (int mr = 0; mr < 2; ++mr)
                af[mr] = load_frag_a(Asl + (wm * 32 + mr * 16 + l16) * 64, ks * 32, hx);
#pragma unroll
            for (int nc = 0; nc < 4; ++nc)
                bfr[nc] = load_frag_b(Bsl + (wn * 64 + nc * 16 + l16) * 64, ks * 32, hx);
#pragma unroll
            for (int mr = 0; mr < 2; ++mr)
#pragma unroll
                for (int nc = 0; nc < 4; ++nc)
                    c[mr][nc] = wmma_bf16(af[mr], bfr[nc], c[mr][nc]);
        }
    }

    // ---- epilogue ----
#pragma unroll
    for (int mr = 0; mr < 2; ++mr) {
#pragma unroll
        for (int nc = 0; nc < 4; ++nc) {
#pragma unroll
            for (int v = 0; v < 8; ++v) {
                int m = tm + mr * 16 + v + 8 * hx;
                int n = tn + nc * 16 + l16;
                float val = c[mr][nc][v];
                if (EPI == 2) {
                    out32[(size_t)m * N + n] = val;
                } else if (EPI == 0) {
                    int b = m / kS, s = m % kS;
                    int h = n / kHD, hd = n % kHD;
                    out16[(((size_t)b * kH + h) * kHD + hd) * kS + s] = (bf16_t)val;
                } else {
                    int b = m / kS, s = m % kS;
                    int h = n / kHD, hd = n % kHD;
                    float pv = __shfl_xor(val, 1, 32);  // partner lane holds n^1
                    float cv = cosT[s * (kHD / 2) + (hd >> 1)];
                    float sv = sinT[s * (kHD / 2) + (hd >> 1)];
                    float r  = (hd & 1) ? (pv * sv + val * cv)
                                        : (val * cv - pv * sv);
                    out16[(((size_t)b * kH + h) * kS + s) * kHD + hd] = (bf16_t)r;
                }
            }
        }
    }
}

// ---------------------------------------------------------------------------
// Flash attention over one (b,h): Qh/Kh are [b,h,s,hd] bf16, Vt is [b,h,hd,s].
// Block = 128 threads = 4 waves, each wave owns 16 q-rows of a 64-row q tile.
// Output: AO[b,s,h,hd] bf16.
// ---------------------------------------------------------------------------
__global__ __launch_bounds__(128) void flash_attn_kernel(
    const bf16_t* __restrict__ Qh, const bf16_t* __restrict__ Kh,
    const bf16_t* __restrict__ Vt, bf16_t* __restrict__ AO) {
    __shared__ __align__(16) bf16_t plds[4][16 * 64];

    const int lane = threadIdx.x & 31;
    const int w    = threadIdx.x >> 5;
    const int hx   = lane >> 4;
    const int l16  = lane & 15;
    const int qt   = blockIdx.x;   // q tile (64 rows)
    const int bh   = blockIdx.y;   // b*H + h
    const int b    = bh / kH;
    const int h    = bh % kH;
    const int qbase = qt * 64;
    const int qrow0 = qbase + w * 16;

    const bf16_t* Qb = Qh + (size_t)bh * kS * kHD;
    const bf16_t* Kb = Kh + (size_t)bh * kS * kHD;
    const bf16_t* Vb = Vt + (size_t)bh * kHD * kS;

    // Q fragments for the whole HD=128 (4 K-steps of 32), kept in registers.
    v16bf aq[4];
    const bf16_t* qrow = Qb + (size_t)(qrow0 + l16) * kHD;
#pragma unroll
    for (int ks = 0; ks < 4; ++ks) aq[ks] = load_frag_a(qrow, ks * 32, hx);

    v8f o[8] = {};  // 16 rows x 128 cols f32 accumulator
    float rmax[8], rsum[8];
#pragma unroll
    for (int v = 0; v < 8; ++v) { rmax[v] = -1e30f; rsum[v] = 0.0f; }

    const float scale = 0.08838834764831845f;  // 1/sqrt(128)
    const int nkt = qt + 1;                    // causal: only tiles <= qt

    for (int kt = 0; kt < nkt; ++kt) {
        const int kbase = kt * 64;
        const bool diag = (kt == qt);

        // ---- scores: 16x64 = 4 tiles of 16x16, K-loop over HD ----
        v8f sc[4] = {};
#pragma unroll
        for (int nt = 0; nt < 4; ++nt) {
            const bf16_t* kcol = Kb + (size_t)(kbase + nt * 16 + l16) * kHD;
#pragma unroll
            for (int ks = 0; ks < 4; ++ks) {
                v16bf bk = load_frag_b(kcol, ks * 32, hx);
                sc[nt] = wmma_bf16(aq[ks], bk, sc[nt]);
            }
        }

        // ---- online softmax (rows live across lanes 0-15 / 16-31) ----
#pragma unroll
        for (int v = 0; v < 8; ++v) {
            int q = qrow0 + v + 8 * hx;
            float mv = -1e30f;
#pragma unroll
            for (int nt = 0; nt < 4; ++nt) {
                int kk = kbase + nt * 16 + l16;
                float s = sc[nt][v] * scale;
                if (diag && kk > q) s = -1e30f;
                sc[nt][v] = s;
                mv = fmaxf(mv, s);
            }
#pragma unroll
            for (int off = 1; off < 16; off <<= 1)
                mv = fmaxf(mv, __shfl_xor(mv, off, 32));
            float nm   = fmaxf(rmax[v], mv);
            float corr = __expf(rmax[v] - nm);
            rmax[v] = nm;
            float lsum = 0.0f;
#pragma unroll
            for (int nt = 0; nt < 4; ++nt) {
                float e = __expf(sc[nt][v] - nm);
                sc[nt][v] = e;
                lsum += e;
            }
#pragma unroll
            for (int off = 1; off < 16; off <<= 1)
                lsum += __shfl_xor(lsum, off, 32);
            rsum[v] = rsum[v] * corr + lsum;
#pragma unroll
            for (int dt = 0; dt < 8; ++dt) o[dt][v] = o[dt][v] * corr;
        }

        __syncthreads();
        // ---- C-layout probs -> A-layout via LDS (per-wave 16x64 tile) ----
#pragma unroll
        for (int nt = 0; nt < 4; ++nt)
#pragma unroll
            for (int v = 0; v < 8; ++v)
                plds[w][(v + 8 * hx) * 64 + nt * 16 + l16] = (bf16_t)sc[nt][v];
        __syncthreads();

        // ---- O += P @ V  (K = 64 keys -> 2 WMMA K-steps) ----
#pragma unroll
        for (int ks = 0; ks < 2; ++ks) {
            FragBF fa;
            const bf16_t* pp = &plds[w][l16 * 64 + ks * 32];
            fa.q[0] = *(const v4i*)(pp + hx * 8);
            fa.q[1] = *(const v4i*)(pp + 16 + hx * 8);
#pragma unroll
            for (int dt = 0; dt < 8; ++dt) {
                const bf16_t* vcol = Vb + (size_t)(dt * 16 + l16) * kS + kbase;
                v16bf bv = load_frag_b(vcol, ks * 32, hx);
                o[dt] = wmma_bf16(fa.v, bv, o[dt]);
            }
        }
    }

    // ---- normalize + store to [b,s,h,hd] bf16 ----
#pragma unroll
    for (int v = 0; v < 8; ++v) {
        int q = qrow0 + v + 8 * hx;
        float inv = 1.0f / rsum[v];
#pragma unroll
        for (int dt = 0; dt < 8; ++dt) {
            int d = dt * 16 + l16;
            AO[(((size_t)b * kS + q) * kH + h) * kHD + d] = (bf16_t)(o[dt][v] * inv);
        }
    }
}

// ---------------------------------------------------------------------------
extern "C" void kernel_launch(void* const* d_in, const int* in_sizes, int n_in,
                              void* d_out, int out_size, void* d_ws,
                              size_t ws_size, hipStream_t stream) {
    const float* x    = (const float*)d_in[0];
    const float* wq   = (const float*)d_in[1];
    const float* wk   = (const float*)d_in[2];
    const float* wv   = (const float*)d_in[3];
    const float* wo   = (const float*)d_in[4];
    const float* cosT = (const float*)d_in[5];
    const float* sinT = (const float*)d_in[6];
    float* out = (float*)d_out;

    const size_t nx = (size_t)kB * kS * kD;  // 16.78M elements
    const size_t nw = (size_t)kD * kD;       // 16.78M elements

    bf16_t* xb  = (bf16_t*)d_ws;
    bf16_t* wqb = xb + nx;
    bf16_t* wkb = wqb + nw;
    bf16_t* wvb = wkb + nw;
    bf16_t* wob = wvb + nw;
    bf16_t* Qh  = wob + nw;  // [b,h,s,hd]
    bf16_t* Kh  = Qh + nx;   // [b,h,s,hd]
    bf16_t* Vt  = Kh + nx;   // [b,h,hd,s]
    bf16_t* AO  = Vt + nx;   // [b,s,h,hd]

    const int cvtB = 256;
    cvt_f32_bf16<<<(int)((nx + cvtB - 1) / cvtB), cvtB, 0, stream>>>(x, xb, (int)nx);
    cvt_f32_bf16<<<(int)((nw + cvtB - 1) / cvtB), cvtB, 0, stream>>>(wq, wqb, (int)nw);
    cvt_f32_bf16<<<(int)((nw + cvtB - 1) / cvtB), cvtB, 0, stream>>>(wk, wkb, (int)nw);
    cvt_f32_bf16<<<(int)((nw + cvtB - 1) / cvtB), cvtB, 0, stream>>>(wv, wvb, (int)nw);
    cvt_f32_bf16<<<(int)((nw + cvtB - 1) / cvtB), cvtB, 0, stream>>>(wo, wob, (int)nw);

    const int M = kB * kS, N = kD, K = kD;
    dim3 gemmGrid(N / 128, M / 128);  // (32, 32)

    // Q, K projections with fused RoPE -> head-major layout
    gemm_bf16_kernel<1><<<gemmGrid, 256, 0, stream>>>(xb, wqb, Qh, nullptr,
                                                      cosT, sinT, M, N, K);
    gemm_bf16_kernel<1><<<gemmGrid, 256, 0, stream>>>(xb, wkb, Kh, nullptr,
                                                      cosT, sinT, M, N, K);
    // V projection -> transposed per-head layout
    gemm_bf16_kernel<0><<<gemmGrid, 256, 0, stream>>>(xb, wvb, Vt, nullptr,
                                                      cosT, sinT, M, N, K);

    // Flash attention: grid (q-tiles, b*h)
    dim3 agrid(kS / 64, kB * kH);
    flash_attn_kernel<<<agrid, 128, 0, stream>>>(Qh, Kh, Vt, AO);

    // Output projection, f32 result
    gemm_bf16_kernel<2><<<gemmGrid, 256, 0, stream>>>(AO, wob, nullptr, out,
                                                      cosT, sinT, M, N, K);
}
